// MoE_8504035246725
// MI455X (gfx1250) — matrix operand order repaired and checked by reading, not compiled, not run on hardware
//
#include <hip/hip_runtime.h>
#include <math.h>

// ---------------------------------------------------------------------------
// MoE (noisy top-4 gating, 8 experts) for MI455X / gfx1250, wave32 + WMMA.
// N=4096, D=1024, H=4096, E=8. Dense expert compute = 275 GFLOP bf16 via
// v_wmma_f32_16x16x32_bf16. Block tile 64(M) x 128(N), K chunked by 64,
// double-buffered LDS staging through GLOBAL_LOAD_ASYNC_TO_LDS_B128
// (ASYNCcnt), one barrier per stage. W_bf16 (64MB) is L2-resident; B traffic
// is (N/64) * 64MB = 4GB of coalesced L2 reads.
// ---------------------------------------------------------------------------

typedef __attribute__((ext_vector_type(16))) __bf16 v16bf;
typedef __attribute__((ext_vector_type(8)))  __bf16 v8bf;
typedef __attribute__((ext_vector_type(8)))  float  v8f;
typedef __attribute__((ext_vector_type(4)))  int    v4i;

#define N_TOK 4096
#define DIM   1024
#define HID   4096
#define NEXP  8
#define TOPK  4

#define BM 64                       // tokens per block
#define BN 128                      // H columns per block (8 waves x 16)
#define BK 64                       // K elements staged per stage
#define NSTAGE (NEXP * (DIM / BK))  // 8 experts x 16 k-chunks = 128 stages

#define PX 144                      // x stage row pitch (128B data + 16B pad)
#define PB 272                      // B stage row pitch (256B data + 16B pad)
#define XBYTES (BM * PX)            // 9216
#define BBYTES (BK * PB)            // 17408

// ---- CDNA5 async global->LDS path (ASYNCcnt), with sync fallback ----------
#if defined(__has_builtin)
#  if __has_builtin(__builtin_amdgcn_global_load_async_to_lds_b128)
#    define HAVE_ASYNC_LDS 1
#  endif
#endif
#ifndef HAVE_ASYNC_LDS
#  define HAVE_ASYNC_LDS 0
#endif

#if defined(__has_builtin) && __has_builtin(__builtin_amdgcn_s_wait_asynccnt)
#  define WAIT_ASYNC(n) __builtin_amdgcn_s_wait_asynccnt(n)
#else
#  define WAIT_ASYNC(n) asm volatile("s_wait_asynccnt %0" ::"n"(n) : "memory")
#endif

#if HAVE_ASYNC_LDS
__device__ __forceinline__ void cp16_async(const void* g, void* l) {
    __builtin_amdgcn_global_load_async_to_lds_b128(
        (__attribute__((address_space(1))) v4i*)g,
        (__attribute__((address_space(3))) v4i*)l, 0, 0);
}
#endif

__device__ __forceinline__ unsigned short f2bf_rne(float f) {
    unsigned u = __float_as_uint(f);
    unsigned r = 0x7FFFu + ((u >> 16) & 1u);   // round-to-nearest-even
    return (unsigned short)((u + r) >> 16);
}

// ---------------------------------------------------------------------------
// Kernel 1: noisy top-k gating. One wave per token; w_gate/w_noise (64KB)
// cached in LDS. gates[N,E] dense fp32 written to workspace.
// ---------------------------------------------------------------------------
__global__ __launch_bounds__(256) void moe_gate_kernel(
    const float* __restrict__ x, const float* __restrict__ noise,
    const float* __restrict__ w_gate, const float* __restrict__ w_noise,
    float* __restrict__ gates)
{
    __shared__ float s_wg[DIM * NEXP];
    __shared__ float s_wn[DIM * NEXP];

    const int tid = threadIdx.x;
    {
        const float4* wg4 = (const float4*)w_gate;
        const float4* wn4 = (const float4*)w_noise;
        float4* swg4 = (float4*)s_wg;
        float4* swn4 = (float4*)s_wn;
        for (int i = tid; i < DIM * NEXP / 4; i += 256) {
            swg4[i] = wg4[i];
            swn4[i] = wn4[i];
        }
    }
    __syncthreads();

    const int wave = tid >> 5;
    const int lane = tid & 31;
    const int n = blockIdx.x * 8 + wave;

    float ag[NEXP], an[NEXP];
#pragma unroll
    for (int e = 0; e < NEXP; ++e) { ag[e] = 0.0f; an[e] = 0.0f; }

    const float* xrow = x + (size_t)n * DIM;
    for (int d = lane; d < DIM; d += 32) {
        float xv = xrow[d];
        const float4* g4 = (const float4*)&s_wg[d * NEXP];
        const float4* q4 = (const float4*)&s_wn[d * NEXP];
        float4 g0 = g4[0], g1 = g4[1];
        float4 q0 = q4[0], q1 = q4[1];
        ag[0] += xv * g0.x; ag[1] += xv * g0.y; ag[2] += xv * g0.z; ag[3] += xv * g0.w;
        ag[4] += xv * g1.x; ag[5] += xv * g1.y; ag[6] += xv * g1.z; ag[7] += xv * g1.w;
        an[0] += xv * q0.x; an[1] += xv * q0.y; an[2] += xv * q0.z; an[3] += xv * q0.w;
        an[4] += xv * q1.x; an[5] += xv * q1.y; an[6] += xv * q1.z; an[7] += xv * q1.w;
    }
#pragma unroll
    for (int e = 0; e < NEXP; ++e) {
#pragma unroll
        for (int off = 16; off > 0; off >>= 1) {
            ag[e] += __shfl_down(ag[e], off, 32);
            an[e] += __shfl_down(an[e], off, 32);
        }
    }

    if (lane == 0) {
        float logit[NEXP];
#pragma unroll
        for (int e = 0; e < NEXP; ++e) {
            float z = an[e];
            float sp = (z > 20.0f) ? z : log1pf(expf(z));      // softplus
            logit[e] = ag[e] + noise[(size_t)n * NEXP + e] * (sp + 1e-2f);
        }
        float l[NEXP];
#pragma unroll
        for (int e = 0; e < NEXP; ++e) l[e] = logit[e];
        int   idx[TOPK];
        float val[TOPK];
#pragma unroll
        for (int k = 0; k < TOPK; ++k) {   // exact top-4, ties -> lowest index
            float best = -3.402823466e+38f;
            int bi = 0;
#pragma unroll
            for (int e = 0; e < NEXP; ++e)
                if (l[e] > best) { best = l[e]; bi = e; }
            idx[k] = bi; val[k] = best;
            l[bi] = -3.402823466e+38f;
        }
        float mx = val[0], w[TOPK], sum = 0.0f;
#pragma unroll
        for (int k = 0; k < TOPK; ++k) { w[k] = expf(val[k] - mx); sum += w[k]; }
        float inv = 1.0f / sum;
        float g[NEXP];
#pragma unroll
        for (int e = 0; e < NEXP; ++e) g[e] = 0.0f;
#pragma unroll
        for (int k = 0; k < TOPK; ++k) g[idx[k]] = w[k] * inv;
#pragma unroll
        for (int e = 0; e < NEXP; ++e) gates[(size_t)n * NEXP + e] = g[e];
    }
}

// ---------------------------------------------------------------------------
// Kernel 2: fp32 -> bf16 (RNE), float4 in / ushort4 out.
// ---------------------------------------------------------------------------
__global__ __launch_bounds__(256) void cvt_f32_bf16_kernel(
    const float* __restrict__ src, unsigned short* __restrict__ dst, long long n4)
{
    long long i = (long long)blockIdx.x * blockDim.x + threadIdx.x;
    long long stride = (long long)gridDim.x * blockDim.x;
    const float4* s4 = (const float4*)src;
    ushort4* d4 = (ushort4*)dst;
    for (; i < n4; i += stride) {
        float4 v = s4[i];
        ushort4 o;
        o.x = f2bf_rne(v.x); o.y = f2bf_rne(v.y);
        o.z = f2bf_rne(v.z); o.w = f2bf_rne(v.w);
        d4[i] = o;
    }
}

// ---------------------------------------------------------------------------
// Kernel 3: expert GEMMs + gate-weighted combine.
// Grid (HID/BN, N_TOK/BM), 256 threads = 8 waves. Wave w owns columns
// h0+16w..+15 for all 64 tokens: 4 stacked 16x16 accumulators sharing one B
// fragment per WMMA. Per stage all 10 fragments (2 B + 8 A) are preloaded
// into registers, then 8 WMMAs issue as a burst. One barrier per stage:
//   wait_asynccnt(0) -> barrier -> issue(s+1) -> compute(s)
// proves batch-s visibility AND that buffer (s+1)&1's readers are done.
// ---------------------------------------------------------------------------
__global__ __launch_bounds__(256) void moe_expert_kernel(
    const unsigned short* __restrict__ xb,      // [N, D]     bf16
    const unsigned short* __restrict__ wbm,     // [E, D, H]  bf16
    const float* __restrict__ gates,            // [N, E]
    const float* __restrict__ b_experts,        // [E, H]
    float* __restrict__ out)                    // [N, H]
{
    __shared__ unsigned char s_x[2 * XBYTES];   // 18 KB
    __shared__ unsigned char s_b[2 * BBYTES];   // 34 KB
    __shared__ float s_g[BM * NEXP];            // 2 KB

    const int tid  = threadIdx.x;
    const int lane = tid & 31;
    const int wave = tid >> 5;
    const int n0 = blockIdx.y * BM;
    const int h0 = blockIdx.x * BN;

    // gates tile (64 tokens x 8 experts)
    s_g[tid]       = gates[(size_t)(n0 + (tid >> 3)) * NEXP + (tid & 7)];
    s_g[tid + 256] = gates[(size_t)(n0 + 32 + (tid >> 3)) * NEXP + (tid & 7)];

    // ---- per-thread staging geometry ----
    // x chunk: 64 rows x 128B; thread t copies 2x16B: row = t/4, segs 2t%8, +1
    const int xrow = tid >> 2;
    const int xseg = (tid & 3) * 2;
    // B chunk: 64 rows x 256B; thread t copies 4x16B: chunks t + 256*j
    const int brow = tid >> 4;
    const int bseg = tid & 15;

#define STAGE_ADDRS(s, e, kk, lx, lb)                                          \
    const int e = (s) >> 4;                                                    \
    const int kk = ((s) & 15) * BK;                                            \
    unsigned char* lx = s_x + ((s) & 1) * XBYTES;                              \
    unsigned char* lb = s_b + ((s) & 1) * BBYTES;

#if HAVE_ASYNC_LDS
#define ISSUE_STAGE(s)                                                         \
    do {                                                                       \
        STAGE_ADDRS(s, e_, kk_, lx_, lb_)                                      \
        const unsigned short* gx = xb + (size_t)(n0 + xrow) * DIM + kk_ + xseg * 8; \
        cp16_async(gx,     lx_ + xrow * PX + xseg * 16);                       \
        cp16_async(gx + 8, lx_ + xrow * PX + xseg * 16 + 16);                  \
        const unsigned short* gb = wbm +                                       \
            ((size_t)e_ * DIM + kk_ + brow) * HID + h0 + bseg * 8;             \
        cp16_async(gb,                    lb_ + brow * PB + bseg * 16);        \
        cp16_async(gb + (size_t)16 * HID, lb_ + (brow + 16) * PB + bseg * 16); \
        cp16_async(gb + (size_t)32 * HID, lb_ + (brow + 32) * PB + bseg * 16); \
        cp16_async(gb + (size_t)48 * HID, lb_ + (brow + 48) * PB + bseg * 16); \
    } while (0)
#else
#define ISSUE_STAGE(s)                                                         \
    do {                                                                       \
        STAGE_ADDRS(s, e_, kk_, lx_, lb_)                                      \
        const unsigned short* gx = xb + (size_t)(n0 + xrow) * DIM + kk_ + xseg * 8; \
        *(uint4*)(lx_ + xrow * PX + xseg * 16)      = *(const uint4*)gx;       \
        *(uint4*)(lx_ + xrow * PX + xseg * 16 + 16) = *(const uint4*)(gx + 8); \
        const unsigned short* gb = wbm +                                       \
            ((size_t)e_ * DIM + kk_ + brow) * HID + h0 + bseg * 8;             \
        *(uint4*)(lb_ + brow * PB + bseg * 16)        = *(const uint4*)gb;     \
        *(uint4*)(lb_ + (brow + 16) * PB + bseg * 16) = *(const uint4*)(gb + (size_t)16 * HID); \
        *(uint4*)(lb_ + (brow + 32) * PB + bseg * 16) = *(const uint4*)(gb + (size_t)32 * HID); \
        *(uint4*)(lb_ + (brow + 48) * PB + bseg * 16) = *(const uint4*)(gb + (size_t)48 * HID); \
    } while (0)
#endif

    // ---- compute-side geometry (WMMA fragment layouts per CDNA5 ISA) ----
    const int arow  = lane & 15;            // M row within 16-row fragment
    const int koff  = (lane >> 4) * 8;      // A-frag K sub-offset (0 or 8)
    const int mhalf = (lane >> 4) << 3;     // C-frag M offset (0 or 8)
    const int outcol = h0 + wave * 16 + arow;

    v8f acc[4]  = {};                       // per-expert, 4 M-fragments
    v8f oacc[4] = {};                       // gate-weighted running output

#if HAVE_ASYNC_LDS
    ISSUE_STAGE(0);
#endif

    for (int s = 0; s < NSTAGE; ++s) {
#if HAVE_ASYNC_LDS
        WAIT_ASYNC(0);                      // own batch-s copies have landed
        __syncthreads();                    // => batch s visible everywhere,
                                            //    and stage s-1 readers done
        if (s + 1 < NSTAGE) ISSUE_STAGE(s + 1);
#else
        __syncthreads();                    // readers of this buffer done
        ISSUE_STAGE(s);
        __syncthreads();
#endif
        {
            STAGE_ADDRS(s, e_, kk_, lx_, lb_)
            (void)kk_;
            // Preload every fragment of the stage, then burst 8 WMMAs.
            v16bf bfr[2];
            v16bf afr[2][4];
#pragma unroll
            for (int kc = 0; kc < 2; ++kc) {
                // B fragment: lane holds col wave*16+arow, K in table order (matches A)
#pragma unroll
                for (int i = 0; i < 16; ++i)
                    bfr[kc][i] = *(const __bf16*)(lb_ + (kc * 32 + koff + i + (i >= 8 ? 8 : 0)) * PB + (wave * 16 + arow) * 2);
#pragma unroll
                for (int mf = 0; mf < 4; ++mf) {
                    const unsigned char* ap =
                        lx_ + (mf * 16 + arow) * PX + kc * 64 + koff * 2;
                    v8bf alo = *(const v8bf*)ap;
                    v8bf ahi = *(const v8bf*)(ap + 32);
#pragma unroll
                    for (int i = 0; i < 8; ++i) {
                        afr[kc][mf][i]     = alo[i];
                        afr[kc][mf][8 + i] = ahi[i];
                    }
                }
            }
#pragma unroll
            for (int kc = 0; kc < 2; ++kc)
#pragma unroll
                for (int mf = 0; mf < 4; ++mf)
                    acc[mf] = __builtin_amdgcn_wmma_f32_16x16x32_bf16(
                        false, afr[kc][mf], false, bfr[kc],
                        (short)0, acc[mf], false, false);

            if ((s & 15) == 15) {           // expert finished: fold with gate+bias
                float bias = b_experts[e_ * HID + outcol];
#pragma unroll
                for (int mf = 0; mf < 4; ++mf) {
#pragma unroll
                    for (int r = 0; r < 8; ++r) {
                        float g = s_g[(mf * 16 + mhalf + r) * NEXP + e_];
                        oacc[mf][r] += g * (acc[mf][r] + bias);
                    }
                    acc[mf] = (v8f){};
                }
            }
        }
    }

#pragma unroll
    for (int mf = 0; mf < 4; ++mf)
#pragma unroll
        for (int r = 0; r < 8; ++r)
            out[(size_t)(n0 + mf * 16 + mhalf + r) * HID + outcol] = oacc[mf][r];
}

// ---------------------------------------------------------------------------
// Launch. Workspace: gates (128KB) | x_bf16 (8MB) | W_bf16 (64MB) ~= 72.2MB.
// ---------------------------------------------------------------------------
extern "C" void kernel_launch(void* const* d_in, const int* in_sizes, int n_in,
                              void* d_out, int out_size, void* d_ws, size_t ws_size,
                              hipStream_t stream)
{
    (void)in_sizes; (void)n_in; (void)out_size; (void)ws_size;
    const float* x       = (const float*)d_in[0];
    const float* noise   = (const float*)d_in[1];
    const float* w_gate  = (const float*)d_in[2];
    const float* w_noise = (const float*)d_in[3];
    const float* W_exp   = (const float*)d_in[4];
    const float* b_exp   = (const float*)d_in[5];
    float* out = (float*)d_out;

    char* ws = (char*)d_ws;
    float*          gates = (float*)ws;                                   // 128 KB
    unsigned short* xbuf  = (unsigned short*)(ws + (128u << 10));          // 8 MB
    unsigned short* wbuf  = (unsigned short*)(ws + (128u << 10) +
                                              (size_t)N_TOK * DIM * 2);    // 64 MB

    moe_gate_kernel<<<N_TOK / 8, 256, 0, stream>>>(x, noise, w_gate, w_noise, gates);
    cvt_f32_bf16_kernel<<<2048, 256, 0, stream>>>(x, xbuf,
        (long long)((size_t)N_TOK * DIM / 4));
    cvt_f32_bf16_kernel<<<8192, 256, 0, stream>>>(W_exp, wbuf,
        (long long)((size_t)NEXP * DIM * HID / 4));

    dim3 grid(HID / BN, N_TOK / BM);
    moe_expert_kernel<<<grid, 256, 0, stream>>>(xbuf, wbuf, gates, b_exp, out);
}
